// FlashDeepseekLayer_39204461478520
// MI455X (gfx1250) — compile-verified, matrix operand order checked by reading, not run on hardware
//
#include <hip/hip_runtime.h>
#include <math.h>

// ---- problem constants (match reference) ----
#define T_TOK   1024
#define H_DIM   2048
#define E_EXP   16
#define K_SEL   4
#define I_DIM   1408
#define IS_DIM  2816
#define N_GROUP 4
#define G_SIZE  (E_EXP / N_GROUP)   // 4
#define R_SCALE 2.5f

typedef __attribute__((ext_vector_type(16))) __bf16 v16bf;
typedef __attribute__((ext_vector_type(8)))  float  v8f;

// One WMMA 16-bit operand fragment: 8 VGPRs = 2x uint4 (b128) loads.
union Frag { v16bf v; uint4 q[2]; };

__device__ __forceinline__ float fast_sigmoid(float x) {
  return __builtin_amdgcn_rcpf(1.0f + __expf(-x));
}

#define WMMA_BF16(A, B, C) \
  __builtin_amdgcn_wmma_f32_16x16x32_bf16(false, (A), false, (B), (short)0, (C), false, false)

// ---------------------------------------------------------------------------
// 1) Router: DeepSeek-V3 noaux_tc, one block per token.
// ---------------------------------------------------------------------------
__global__ void route_kernel(const float* __restrict__ x,
                             const float* __restrict__ gw,
                             const float* __restrict__ gb,
                             float* __restrict__ cw) {
  __shared__ float part[256];
  __shared__ float scores[E_EXP];
  const int t = blockIdx.x;
  const int e = threadIdx.x >> 4;     // 16 threads per expert
  const int j = threadIdx.x & 15;
  const float* xr = x + (size_t)t * H_DIM;
  const float* gr = gw + (size_t)e * H_DIM;
  float s = 0.f;
  for (int h = j; h < H_DIM; h += 16) s += xr[h] * gr[h];
  part[threadIdx.x] = s;
  __syncthreads();
  if (threadIdx.x < E_EXP) {
    float l = 0.f;
    for (int k = 0; k < 16; ++k) l += part[threadIdx.x * 16 + k];
    scores[threadIdx.x] = fast_sigmoid(l);
  }
  __syncthreads();
  if (threadIdx.x == 0) {
    float biased[E_EXP];
    for (int i = 0; i < E_EXP; ++i) biased[i] = scores[i] + gb[i];
    float gs[N_GROUP];
    for (int g = 0; g < N_GROUP; ++g) {         // sum of top-2 biased per group
      float m1 = -1e30f, m2 = -1e30f;
      for (int k = 0; k < G_SIZE; ++k) {
        float v = biased[g * G_SIZE + k];
        if (v > m1) { m2 = m1; m1 = v; } else if (v > m2) m2 = v;
      }
      gs[g] = m1 + m2;
    }
    int g1 = 0;
    for (int g = 1; g < N_GROUP; ++g) if (gs[g] > gs[g1]) g1 = g;
    int g2 = -1;
    for (int g = 0; g < N_GROUP; ++g) {
      if (g == g1) continue;
      if (g2 < 0 || gs[g] > gs[g2]) g2 = g;
    }
    float masked[E_EXP];
    for (int i = 0; i < E_EXP; ++i) {
      int g = i / G_SIZE;
      masked[i] = (g == g1 || g == g2) ? biased[i] : 0.0f;
    }
    int idx[K_SEL]; float w[K_SEL]; float wsum = 0.f;
    bool used[E_EXP] = {false};
    for (int k = 0; k < K_SEL; ++k) {           // top-k, lowest-index tiebreak
      int best = -1;
      for (int i = 0; i < E_EXP; ++i) {
        if (used[i]) continue;
        if (best < 0 || masked[i] > masked[best]) best = i;
      }
      used[best] = true; idx[k] = best; w[k] = scores[best]; wsum += w[k];
    }
    float inv = R_SCALE * __builtin_amdgcn_rcpf(wsum + 1e-20f);
    float* outr = cw + (size_t)t * E_EXP;
    for (int i = 0; i < E_EXP; ++i) outr[i] = 0.f;
    for (int k = 0; k < K_SEL; ++k) outr[idx[k]] += w[k] * inv;
  }
}

// ---------------------------------------------------------------------------
// 2) fp32 -> bf16 cast of hidden states
// ---------------------------------------------------------------------------
__global__ void xcast_kernel(const float* __restrict__ x,
                             __bf16* __restrict__ xb, int n) {
  int i = blockIdx.x * blockDim.x + threadIdx.x;
  if (i < n) xb[i] = (__bf16)x[i];
}

// ---------------------------------------------------------------------------
// 3) Transpose + cast: in fp32 [B][R][C] -> out bf16 [B][C][R].
//    32x32 tiles via LDS; both global sides coalesced. 256 threads (32x8).
// ---------------------------------------------------------------------------
__global__ void transpose_cast_kernel(const float* __restrict__ in,
                                      __bf16* __restrict__ out,
                                      int R, int C) {
  __shared__ __bf16 tile[32][33];
  const int b = blockIdx.z;
  const float* inp = in + (size_t)b * R * C;
  __bf16* outp = out + (size_t)b * R * C;
  const int c0 = blockIdx.x * 32, r0 = blockIdx.y * 32;
  const int tx = threadIdx.x & 31, ty = threadIdx.x >> 5;
#pragma unroll
  for (int i = 0; i < 32; i += 8)
    tile[ty + i][tx] = (__bf16)inp[(size_t)(r0 + ty + i) * C + (c0 + tx)];
  __syncthreads();
#pragma unroll
  for (int i = 0; i < 32; i += 8)
    outp[(size_t)(c0 + ty + i) * R + (r0 + tx)] = tile[tx][ty + i];
}

// ---------------------------------------------------------------------------
// 4) SwiGLU activation GEMMs, all-bf16 operands via b128 fragment loads.
//    Wave -> 32 tokens x 32 cols for BOTH gate & up:
//      2 A fragments + 4 B fragments -> 8 WMMAs per k-step (1.5 loads/WMMA).
//    grid: (Idim/128, T/32, nExperts); 128 threads = 4 waves.
//    WgT/WuT pre-transposed: [nE][Idim][H] bf16 (row n contiguous over k).
// ---------------------------------------------------------------------------
template <bool USE_CW>
__global__ void act_kernel(const __bf16* __restrict__ xb,    // [T][H]
                           const __bf16* __restrict__ WgT,   // [nE][Idim][H]
                           const __bf16* __restrict__ WuT,   // [nE][Idim][H]
                           const float* __restrict__ cw,     // [T][E] (if USE_CW)
                           __bf16* __restrict__ out,         // [nE][T][Idim]
                           int Idim) {
  const int lane = threadIdx.x & 31;
  const int wave = threadIdx.x >> 5;
  const int e    = blockIdx.z;
  const int n0   = blockIdx.x * 128 + wave * 32;
  const int t0   = blockIdx.y * 32;
  const int m    = lane & 15;
  const int half = lane >> 4;
  const int nA   = n0 + m;

  // ISA 16-bit fragment: VGPR0-3 = K[8*half .. +8), VGPR4-7 = K[16+8*half .. +8)
  const __bf16* pa = xb + (size_t)(t0 + m) * H_DIM + 8 * half;
  const __bf16* pg = WgT + ((size_t)e * Idim + nA) * H_DIM + 8 * half;
  const __bf16* pu = WuT + ((size_t)e * Idim + nA) * H_DIM + 8 * half;

  v8f cg[2][2] = {};   // [token-subtile][col-subtile]
  v8f cu[2][2] = {};
  for (int k0 = 0; k0 < H_DIM; k0 += 32) {
    Frag a0, a1, bg0, bg1, bu0, bu1;
    a0.q[0]  = *(const uint4*)(pa);
    a0.q[1]  = *(const uint4*)(pa + 16);
    a1.q[0]  = *(const uint4*)(pa + 16 * H_DIM);
    a1.q[1]  = *(const uint4*)(pa + 16 * H_DIM + 16);
    bg0.q[0] = *(const uint4*)(pg);
    bg0.q[1] = *(const uint4*)(pg + 16);
    bg1.q[0] = *(const uint4*)(pg + 16 * H_DIM);
    bg1.q[1] = *(const uint4*)(pg + 16 * H_DIM + 16);
    bu0.q[0] = *(const uint4*)(pu);
    bu0.q[1] = *(const uint4*)(pu + 16);
    bu1.q[0] = *(const uint4*)(pu + 16 * H_DIM);
    bu1.q[1] = *(const uint4*)(pu + 16 * H_DIM + 16);
    cg[0][0] = WMMA_BF16(a0.v, bg0.v, cg[0][0]);
    cg[1][0] = WMMA_BF16(a1.v, bg0.v, cg[1][0]);
    cg[0][1] = WMMA_BF16(a0.v, bg1.v, cg[0][1]);
    cg[1][1] = WMMA_BF16(a1.v, bg1.v, cg[1][1]);
    cu[0][0] = WMMA_BF16(a0.v, bu0.v, cu[0][0]);
    cu[1][0] = WMMA_BF16(a1.v, bu0.v, cu[1][0]);
    cu[0][1] = WMMA_BF16(a0.v, bu1.v, cu[0][1]);
    cu[1][1] = WMMA_BF16(a1.v, bu1.v, cu[1][1]);
    pa += 32; pg += 32; pu += 32;
  }

  __bf16* op = out + (size_t)e * T_TOK * Idim;
#pragma unroll
  for (int ts = 0; ts < 2; ++ts) {
#pragma unroll
    for (int r = 0; r < 8; ++r) {
      const int t = t0 + ts * 16 + r + 8 * half;  // D: VGPR r -> M = r + 8*half
      float sc = 1.0f;
      if (USE_CW) sc = cw[t * E_EXP + e];
      float g0 = cg[ts][0][r], g1 = cg[ts][1][r];
      float a0 = g0 * fast_sigmoid(g0) * cu[ts][0][r] * sc;
      float a1 = g1 * fast_sigmoid(g1) * cu[ts][1][r] * sc;
      op[(size_t)t * Idim + nA]      = (__bf16)a0;
      op[(size_t)t * Idim + nA + 16] = (__bf16)a1;
    }
  }
}

// ---------------------------------------------------------------------------
// 5) Down-projection + combine:  out = sum_e (cw_e*act_e) @ wd_e + act_s @ sd
//    Wave -> 32 tokens x 64 cols: 2 A frags + 4 B frags -> 8 WMMAs/k-step.
//    grid: (H/256, T/32); 128 threads = 4 waves.
//    wdT: [E][H][I] bf16; sdT: [H][IS] bf16 (pre-transposed).
// ---------------------------------------------------------------------------
__global__ void down_kernel(const __bf16* __restrict__ sact,   // [E][T][I]
                            const __bf16* __restrict__ wdT,    // [E][H][I]
                            const __bf16* __restrict__ sact_s, // [T][IS]
                            const __bf16* __restrict__ sdT,    // [H][IS]
                            float* __restrict__ out) {         // [T][H]
  const int lane = threadIdx.x & 31;
  const int wave = threadIdx.x >> 5;
  const int n0   = blockIdx.x * 256 + wave * 64;
  const int t0   = blockIdx.y * 32;
  const int m    = lane & 15;
  const int half = lane >> 4;
  const int nA   = n0 + m;

  v8f acc[2][4] = {};   // [token-subtile][col-subtile]

  for (int e = 0; e < E_EXP; ++e) {
    const __bf16* pa = sact + ((size_t)e * T_TOK + (t0 + m)) * I_DIM + 8 * half;
    const __bf16* pb = wdT + ((size_t)e * H_DIM + nA) * I_DIM + 8 * half;
    for (int k0 = 0; k0 < I_DIM; k0 += 32) {
      Frag a0, a1;
      a0.q[0] = *(const uint4*)(pa);
      a0.q[1] = *(const uint4*)(pa + 16);
      a1.q[0] = *(const uint4*)(pa + 16 * I_DIM);
      a1.q[1] = *(const uint4*)(pa + 16 * I_DIM + 16);
#pragma unroll
      for (int s = 0; s < 4; ++s) {
        Frag b;
        b.q[0] = *(const uint4*)(pb + s * 16 * I_DIM);
        b.q[1] = *(const uint4*)(pb + s * 16 * I_DIM + 16);
        acc[0][s] = WMMA_BF16(a0.v, b.v, acc[0][s]);
        acc[1][s] = WMMA_BF16(a1.v, b.v, acc[1][s]);
      }
      pa += 32; pb += 32;
    }
  }
  // shared expert contribution
  {
    const __bf16* pa = sact_s + (size_t)(t0 + m) * IS_DIM + 8 * half;
    const __bf16* pb = sdT + (size_t)nA * IS_DIM + 8 * half;
    for (int k0 = 0; k0 < IS_DIM; k0 += 32) {
      Frag a0, a1;
      a0.q[0] = *(const uint4*)(pa);
      a0.q[1] = *(const uint4*)(pa + 16);
      a1.q[0] = *(const uint4*)(pa + 16 * IS_DIM);
      a1.q[1] = *(const uint4*)(pa + 16 * IS_DIM + 16);
#pragma unroll
      for (int s = 0; s < 4; ++s) {
        Frag b;
        b.q[0] = *(const uint4*)(pb + s * 16 * IS_DIM);
        b.q[1] = *(const uint4*)(pb + s * 16 * IS_DIM + 16);
        acc[0][s] = WMMA_BF16(a0.v, b.v, acc[0][s]);
        acc[1][s] = WMMA_BF16(a1.v, b.v, acc[1][s]);
      }
      pa += 32; pb += 32;
    }
  }
#pragma unroll
  for (int ts = 0; ts < 2; ++ts) {
#pragma unroll
    for (int s = 0; s < 4; ++s) {
#pragma unroll
      for (int r = 0; r < 8; ++r) {
        const int t = t0 + ts * 16 + r + 8 * half;
        out[(size_t)t * H_DIM + nA + 16 * s] = acc[ts][s][r];
      }
    }
  }
}

// ---------------------------------------------------------------------------
extern "C" void kernel_launch(void* const* d_in, const int* in_sizes, int n_in,
                              void* d_out, int out_size, void* d_ws, size_t ws_size,
                              hipStream_t stream) {
  (void)in_sizes; (void)n_in; (void)out_size; (void)ws_size;
  const float* x  = (const float*)d_in[0];
  const float* gw = (const float*)d_in[1];
  const float* gb = (const float*)d_in[2];
  const float* wg = (const float*)d_in[3];
  const float* wu = (const float*)d_in[4];
  const float* wd = (const float*)d_in[5];
  const float* sg = (const float*)d_in[6];
  const float* su = (const float*)d_in[7];
  const float* sd = (const float*)d_in[8];
  float* out = (float*)d_out;

  // workspace layout
  char* p = (char*)d_ws;
  float* cw      = (float*)p;      p += 65536;
  __bf16* xb     = (__bf16*)p;     p += (size_t)T_TOK * H_DIM * 2;
  __bf16* sact   = (__bf16*)p;     p += (size_t)E_EXP * T_TOK * I_DIM * 2;
  __bf16* sact_s = (__bf16*)p;     p += (size_t)T_TOK * IS_DIM * 2;
  __bf16* wgT    = (__bf16*)p;     p += (size_t)E_EXP * H_DIM * I_DIM * 2;
  __bf16* wuT    = (__bf16*)p;     p += (size_t)E_EXP * H_DIM * I_DIM * 2;
  __bf16* wdT    = (__bf16*)p;     p += (size_t)E_EXP * H_DIM * I_DIM * 2;
  __bf16* sgT    = (__bf16*)p;     p += (size_t)H_DIM * IS_DIM * 2;
  __bf16* suT    = (__bf16*)p;     p += (size_t)H_DIM * IS_DIM * 2;
  __bf16* sdT    = (__bf16*)p;     p += (size_t)H_DIM * IS_DIM * 2;

  route_kernel<<<T_TOK, 256, 0, stream>>>(x, gw, gb, cw);
  xcast_kernel<<<(T_TOK * H_DIM) / 256, 256, 0, stream>>>(x, xb, T_TOK * H_DIM);

  // weight transposes (fp32 [R][C] -> bf16 [C][R])
  transpose_cast_kernel<<<dim3(I_DIM / 32, H_DIM / 32, E_EXP), 256, 0, stream>>>(
      wg, wgT, H_DIM, I_DIM);
  transpose_cast_kernel<<<dim3(I_DIM / 32, H_DIM / 32, E_EXP), 256, 0, stream>>>(
      wu, wuT, H_DIM, I_DIM);
  transpose_cast_kernel<<<dim3(H_DIM / 32, I_DIM / 32, E_EXP), 256, 0, stream>>>(
      wd, wdT, I_DIM, H_DIM);
  transpose_cast_kernel<<<dim3(IS_DIM / 32, H_DIM / 32, 1), 256, 0, stream>>>(
      sg, sgT, H_DIM, IS_DIM);
  transpose_cast_kernel<<<dim3(IS_DIM / 32, H_DIM / 32, 1), 256, 0, stream>>>(
      su, suT, H_DIM, IS_DIM);
  transpose_cast_kernel<<<dim3(H_DIM / 32, IS_DIM / 32, 1), 256, 0, stream>>>(
      sd, sdT, IS_DIM, H_DIM);

  act_kernel<true><<<dim3(I_DIM / 128, T_TOK / 32, E_EXP), 128, 0, stream>>>(
      xb, wgT, wuT, cw, sact, I_DIM);
  act_kernel<false><<<dim3(IS_DIM / 128, T_TOK / 32, 1), 128, 0, stream>>>(
      xb, sgT, suT, nullptr, sact_s, IS_DIM);
  down_kernel<<<dim3(H_DIM / 256, T_TOK / 32), 128, 0, stream>>>(
      sact, wdT, sact_s, sdT, out);
}
